// Encoder_Decoder_30580167147779
// MI455X (gfx1250) — compile-verified, matrix operand order checked by reading, not run on hardware
//
#include <hip/hip_runtime.h>
#include <hip/hip_bf16.h>

#define NCLS 80
#define SEG  512
#define TLEN 513
#define HID  128
#define NTOT (NCLS*SEG)

typedef __attribute__((ext_vector_type(16))) __bf16 v16bf;
typedef __attribute__((ext_vector_type(8)))  float  v8f;

union FragU { v16bf bf; uint4 q[2]; unsigned short h[16]; };

__device__ __forceinline__ unsigned short f2bf(float f) {
  unsigned int u = __float_as_uint(f);
  unsigned int r = (u + 0x7FFFu + ((u >> 16) & 1u)) >> 16;
  return (unsigned short)r;
}
// permute k within a 32-wide k-step so each lane's fragment is contiguous:
// groups of 8: {0..7}->0, {8..15}->16, {16..23}->8, {24..31}->24  (swap bits 3,4)
__device__ __forceinline__ int kperm(int k) {
  return (k & 7) | ((k & 8) << 1) | ((k & 16) >> 1);
}
__device__ __forceinline__ float sigm_(float x) { return 1.0f / (1.0f + __expf(-x)); }

__device__ __forceinline__ void pack4(unsigned short* dst, float a, float b, float c, float d) {
  unsigned int u0 = (unsigned)f2bf(a) | ((unsigned)f2bf(b) << 16);
  unsigned int u1 = (unsigned)f2bf(c) | ((unsigned)f2bf(d) << 16);
  *(uint2*)dst = make_uint2(u0, u1);
}

__device__ __forceinline__ FragU lds_frag(const unsigned short* base, int row0, int lane) {
  FragU f;
  int m = lane & 15, hh = lane >> 4;
  const uint4* p = (const uint4*)(base + (row0 + m) * 40 + hh * 16);
  f.q[0] = p[0]; f.q[1] = p[1];
  return f;
}

// ---- CDNA5 async global->LDS copy (ASYNCcnt-tracked), CDNA5 ISA ----
// lds_off: wave-relative LDS byte offset in a VGPR; gaddr: 64-bit global address.
__device__ __forceinline__ void async_g2l_b32(unsigned lds_off, const void* gaddr) {
  asm volatile("global_load_async_to_lds_b32 %0, %1, off"
               :: "v"(lds_off), "v"(gaddr) : "memory");
}
__device__ __forceinline__ void wait_async_le1() {
  asm volatile("s_wait_asynccnt 0x1" ::: "memory");
}
__device__ __forceinline__ void wait_async_le0() {
  asm volatile("s_wait_asynccnt 0x0" ::: "memory");
}
__device__ __forceinline__ unsigned lds_offset_of(const void* p) {
  // generic LDS pointers carry the wave-relative LDS offset in the low 32 bits
  return (unsigned)(size_t)p;
}

// ---------------- GEMM 1: x = relu(feat @ appear_W^T + b), K=1024, out bf16 ----------
__global__ __launch_bounds__(256) void k_gemm_feat(
    const float* __restrict__ feat, const float* __restrict__ W,
    const float* __restrict__ bias, unsigned short* __restrict__ xout) {
  __shared__ unsigned short lds_a[128 * 40];
  __shared__ unsigned short lds_b[128 * 40];
  const int tid = threadIdx.x, lane = tid & 31, wave = tid >> 5;
  const int row0 = blockIdx.x * 128;
  v8f acc[8];
#pragma unroll
  for (int j = 0; j < 8; ++j) acc[j] = {};
  for (int kt = 0; kt < 32; ++kt) {
    const int k0 = kt * 32;
#pragma unroll
    for (int it = 0; it < 4; ++it) {
      int g = it * 256 + tid;
      int r = g >> 3, kq = (g & 7) * 4;
      float4 a4 = *(const float4*)(feat + (size_t)(row0 + r) * 1024 + k0 + kq);
      pack4(lds_a + r * 40 + kperm(kq), a4.x, a4.y, a4.z, a4.w);
      float4 b4 = *(const float4*)(W + (size_t)r * 1024 + k0 + kq);
      pack4(lds_b + r * 40 + kperm(kq), b4.x, b4.y, b4.z, b4.w);
    }
    __syncthreads();
    FragU af = lds_frag(lds_a, wave * 16, lane);
#pragma unroll
    for (int jt = 0; jt < 8; ++jt) {
      FragU bf_ = lds_frag(lds_b, jt * 16, lane);
      acc[jt] = __builtin_amdgcn_wmma_f32_16x16x32_bf16(
          false, af.bf, false, bf_.bf, (short)0, acc[jt], false, false);
    }
    __syncthreads();
  }
  const int m = lane & 15, hh = lane >> 4;
#pragma unroll
  for (int jt = 0; jt < 8; ++jt) {
    int col = jt * 16 + m;
    float b = bias[col];
#pragma unroll
    for (int v = 0; v < 8; ++v) {
      int row = row0 + wave * 16 + v + 8 * hh;
      float val = acc[jt][v] + b;
      val = val > 0.f ? val : 0.f;
      xout[(size_t)row * 128 + col] = f2bf(val);
    }
  }
}

// ------- GEMM 2: enc_feat = relu([x|score|box|origin] @ featlin_W^T + b), K=224 -------
__global__ __launch_bounds__(256) void k_gemm_featlin(
    const unsigned short* __restrict__ xin, const float* __restrict__ score,
    const float* __restrict__ box, const float* __restrict__ origin,
    const float* __restrict__ W, const float* __restrict__ bias,
    unsigned short* __restrict__ efout) {
  __shared__ unsigned short lds_a[128 * 40];
  __shared__ unsigned short lds_b[128 * 40];
  const int tid = threadIdx.x, lane = tid & 31, wave = tid >> 5;
  const int row0 = blockIdx.x * 128;
  v8f acc[8];
#pragma unroll
  for (int j = 0; j < 8; ++j) acc[j] = {};
  for (int kt = 0; kt < 7; ++kt) {
    const int k0 = kt * 32;
#pragma unroll
    for (int it = 0; it < 4; ++it) {
      int g = it * 256 + tid;
      int r = g >> 3, kq = (g & 7) * 4;
      int row = row0 + r, k = k0 + kq;
      if (k < 128) {
        uint2 u = *(const uint2*)(xin + (size_t)row * 128 + k);
        *(uint2*)(lds_a + r * 40 + kperm(kq)) = u;
      } else {
        float e[4];
#pragma unroll
        for (int j = 0; j < 4; ++j) {
          int kk = k + j - 128;
          e[j] = (kk == 0) ? score[row]
               : (kk < 5) ? box[(size_t)row * 4 + kk - 1]
                          : origin[(size_t)row * 91 + kk - 5];
        }
        pack4(lds_a + r * 40 + kperm(kq), e[0], e[1], e[2], e[3]);
      }
      float4 b4 = *(const float4*)(W + (size_t)r * 224 + k);
      pack4(lds_b + r * 40 + kperm(kq), b4.x, b4.y, b4.z, b4.w);
    }
    __syncthreads();
    FragU af = lds_frag(lds_a, wave * 16, lane);
#pragma unroll
    for (int jt = 0; jt < 8; ++jt) {
      FragU bf_ = lds_frag(lds_b, jt * 16, lane);
      acc[jt] = __builtin_amdgcn_wmma_f32_16x16x32_bf16(
          false, af.bf, false, bf_.bf, (short)0, acc[jt], false, false);
    }
    __syncthreads();
  }
  const int m = lane & 15, hh = lane >> 4;
#pragma unroll
  for (int jt = 0; jt < 8; ++jt) {
    int col = jt * 16 + m;
    float b = bias[col];
#pragma unroll
    for (int v = 0; v < 8; ++v) {
      int row = row0 + wave * 16 + v + 8 * hh;
      float val = acc[jt][v] + b;
      val = val > 0.f ? val : 0.f;
      efout[(size_t)row * 128 + col] = f2bf(val);
    }
  }
}

// ------ GEMM 3: encoder Gi[c,d] = enc_in(513x128) @ Wih_d[c]^T + bih  (K=128) --------
__global__ __launch_bounds__(256) void k_enc_gi(
    const unsigned short* __restrict__ encf,
    const float* __restrict__ Wih_f, const float* __restrict__ Wih_b,
    const float* __restrict__ bih_f, const float* __restrict__ bih_b,
    float* __restrict__ Gi) {
  extern __shared__ unsigned short eg_lds[];  // 4 ksteps * 384 rows * 40
  const int c = blockIdx.x, d = blockIdx.y;
  const float* Wih = (d ? Wih_b : Wih_f) + (size_t)c * 384 * 128;
  const float* bih = (d ? bih_b : bih_f) + (size_t)c * 384;
  const int tid = threadIdx.x, lane = tid & 31, wave = tid >> 5;
  for (int g = tid; g < 384 * 32; g += 256) {
    int r = g >> 5, kq = (g & 31) * 4;
    float4 w4 = *(const float4*)(Wih + (size_t)r * 128 + kq);
    int kt = kq >> 5;
    pack4(eg_lds + (size_t)kt * 384 * 40 + r * 40 + kperm(kq & 31), w4.x, w4.y, w4.z, w4.w);
  }
  __syncthreads();
  const int m = lane & 15, hh = lane >> 4;
  float* Gb = Gi + (size_t)(c * 2 + d) * TLEN * 384;
  for (int rt = wave; rt < 33; rt += 8) {
    int tr = rt * 16 + m;
    FragU af[4];
    if (tr < 512) {
      const uint4* p = (const uint4*)(encf + ((size_t)(c * 512 + tr)) * 128 + hh * 8);
#pragma unroll
      for (int kt = 0; kt < 4; ++kt) { af[kt].q[0] = p[kt * 4]; af[kt].q[1] = p[kt * 4 + 2]; }
    } else {
      uint4 fill = (tr == 512) ? make_uint4(0x3F803F80u, 0x3F803F80u, 0x3F803F80u, 0x3F803F80u)
                               : make_uint4(0u, 0u, 0u, 0u);
#pragma unroll
      for (int kt = 0; kt < 4; ++kt) { af[kt].q[0] = fill; af[kt].q[1] = fill; }
    }
    for (int jt = 0; jt < 24; ++jt) {
      v8f acc = {};
#pragma unroll
      for (int kt = 0; kt < 4; ++kt) {
        FragU bf_ = lds_frag(eg_lds + (size_t)kt * 384 * 40, jt * 16, lane);
        acc = __builtin_amdgcn_wmma_f32_16x16x32_bf16(
            false, af[kt].bf, false, bf_.bf, (short)0, acc, false, false);
      }
      int col = jt * 16 + m;
      float b = bih[col];
#pragma unroll
      for (int v = 0; v < 8; ++v) {
        int r = rt * 16 + v + 8 * hh;
        if (r < TLEN) Gb[(size_t)r * 384 + col] = acc[v] + b;
      }
    }
  }
}

// ------ GEMM 4: decoder Gi[c] = dec_in(513x32) @ dec_Wih[c]^T + bih  (K=32) ----------
__global__ __launch_bounds__(256) void k_dec_gi(
    const int* __restrict__ labels, const float* __restrict__ emb,
    const float* __restrict__ Wih, const float* __restrict__ bih,
    float* __restrict__ Gi) {
  extern __shared__ unsigned short dg_lds[];  // 768*40 weights + 4*32 emb
  unsigned short* lds_w = dg_lds;
  unsigned short* lds_e = dg_lds + 768 * 40;
  const int c = blockIdx.x;
  const float* Wc = Wih + (size_t)c * 768 * 32;
  const float* bc = bih + (size_t)c * 768;
  const int tid = threadIdx.x, lane = tid & 31, wave = tid >> 5;
  for (int g = tid; g < 768 * 8; g += 256) {
    int r = g >> 3, kq = (g & 7) * 4;
    float4 w4 = *(const float4*)(Wc + (size_t)r * 32 + kq);
    pack4(lds_w + r * 40 + kperm(kq), w4.x, w4.y, w4.z, w4.w);
  }
  if (tid < 128) lds_e[tid] = f2bf(emb[tid]);
  __syncthreads();
  const int m = lane & 15, hh = lane >> 4;
  float* Gb = Gi + (size_t)c * TLEN * 768;
  for (int rt = wave; rt < 33; rt += 8) {
    int tr = rt * 16 + m;
    FragU af;
    if (tr <= 512) {
      int lbl = (tr == 0) ? 2 : labels[c * 512 + tr - 1];
      const uint4* p = (const uint4*)(lds_e + lbl * 32 + hh * 8);
      af.q[0] = p[0]; af.q[1] = p[2];
    } else {
      af.q[0] = af.q[1] = make_uint4(0u, 0u, 0u, 0u);
    }
    for (int jt = 0; jt < 48; ++jt) {
      FragU bf_ = lds_frag(lds_w, jt * 16, lane);
      v8f acc = {};
      acc = __builtin_amdgcn_wmma_f32_16x16x32_bf16(
          false, af.bf, false, bf_.bf, (short)0, acc, false, false);
      int col = jt * 16 + m;
      float b = bc[col];
#pragma unroll
      for (int v = 0; v < 8; ++v) {
        int r = rt * 16 + v + 8 * hh;
        if (r < TLEN) Gb[(size_t)r * 768 + col] = acc[v] + b;
      }
    }
  }
}

// ------------- transpose decoder Whh into [c][k/4][t][4] float4-friendly layout -------
__global__ __launch_bounds__(256) void k_trans_whh(
    const float* __restrict__ Whh, float* __restrict__ WT) {
  size_t g = (size_t)blockIdx.x * 256 + threadIdx.x;
  if (g >= (size_t)NCLS * 768 * 256) return;
  int c = (int)(g / (768 * 256));
  int rem = (int)(g % (768 * 256));
  int t = rem >> 8, k = rem & 255;
  WT[(((size_t)c * 64 + (k >> 2)) * 768 + t) * 4 + (k & 3)] = Whh[g];
}

// --------------- encoder recurrence: Whh (384x128) resident in LDS -------------------
// Gi rows are double-buffered in LDS via CDNA5 async global->LDS prefetch.
__global__ __launch_bounds__(384) void k_enc_rec(
    const float* __restrict__ Gi,
    const float* __restrict__ Whh_f, const float* __restrict__ Whh_b,
    const float* __restrict__ bhh_f, const float* __restrict__ bhh_b,
    float* __restrict__ dech0) {
  extern __shared__ float er_sm[];
  float* WT = er_sm;              // 32*384*4 = 49152 floats (transposed, float4-tiled)
  float* bh = WT + 49152;         // 384
  float* A1 = bh + 384;           // 384
  float* A2 = A1 + 384;           // 128
  float* hx = A2 + 128;           // 128
  float* gbuf = hx + 128;         // 2 * 384 (async Gi double buffer)
  const int c = blockIdx.x, d = blockIdx.y, tid = threadIdx.x;
  const float* Whh = (d ? Whh_b : Whh_f) + (size_t)c * 384 * 128;
  const float* bhh = (d ? bhh_b : bhh_f) + (size_t)c * 384;
  for (int g = tid; g < 384 * 128; g += 384) {
    int r = g >> 7, k = g & 127;
    WT[((k >> 2) * 384 + r) * 4 + (k & 3)] = Whh[g];
  }
  bh[tid] = bhh[tid];
  if (tid < 128) hx[tid] = 0.f;
  __syncthreads();
  const float* Gb = Gi + (size_t)(c * 2 + d) * TLEN * 384;
  const unsigned gb_off = lds_offset_of(gbuf) + (unsigned)tid * 4u;
  {
    int gr0 = d ? 512 : 0;
    async_g2l_b32(gb_off, Gb + (size_t)gr0 * 384 + tid);
  }
  for (int t = 0; t < TLEN; ++t) {
    const int pb = t & 1;
    const bool more = (t + 1 < TLEN);
    if (more) {
      int gr2 = d ? (512 - (t + 1)) : (t + 1);
      async_g2l_b32(gb_off + (unsigned)((pb ^ 1) * 384) * 4u,
                    Gb + (size_t)gr2 * 384 + tid);
    }
    float s = bh[tid];
    const float4* wp = (const float4*)WT + tid;
    const float4* hp = (const float4*)hx;
#pragma unroll 8
    for (int k4 = 0; k4 < 32; ++k4) {
      float4 w = wp[k4 * 384];
      float4 hv = hp[k4];
      s += w.x * hv.x + w.y * hv.y + w.z * hv.z + w.w * hv.w;
    }
    if (more) wait_async_le1(); else wait_async_le0();
    float gi = gbuf[pb * 384 + tid];
    if (tid < 256) A1[tid] = gi + s;
    else { A1[tid] = gi; A2[tid - 256] = s; }
    __syncthreads();
    if (tid < 128) {
      float r = sigm_(A1[tid]);
      float z = sigm_(A1[128 + tid]);
      float n = tanhf(A1[256 + tid] + r * A2[tid]);
      hx[tid] = (1.f - z) * n + z * hx[tid];
    }
    __syncthreads();
  }
  if (tid < 128) dech0[(size_t)c * 256 + d * 128 + tid] = hx[tid];
}

// -------- decoder recurrence: Whh streamed coalesced from L2 (63MB resident) ---------
// Gi rows are double-buffered in LDS via CDNA5 async global->LDS prefetch.
__global__ __launch_bounds__(768) void k_dec_rec(
    const float* __restrict__ Gi, const float* __restrict__ WT,
    const float* __restrict__ bhh, const float* __restrict__ dech0,
    float* __restrict__ hs) {
  __shared__ float A1[768];
  __shared__ float A2[256];
  __shared__ float bh[768];
  __shared__ __align__(16) float hx[256];
  __shared__ float gbuf[2 * 768];
  const int c = blockIdx.x, tid = threadIdx.x;
  bh[tid] = bhh[(size_t)c * 768 + tid];
  if (tid < 256) hx[tid] = dech0[(size_t)c * 256 + tid];
  __syncthreads();
  const float* Gb = Gi + (size_t)c * TLEN * 768;
  const float4* wp = (const float4*)(WT + (size_t)c * 768 * 256) + tid;
  float* hsb = hs + (size_t)c * TLEN * 256;
  const unsigned gb_off = lds_offset_of(gbuf) + (unsigned)tid * 4u;
  async_g2l_b32(gb_off, Gb + tid);
  for (int t = 0; t < TLEN; ++t) {
    const int pb = t & 1;
    const bool more = (t + 1 < TLEN);
    if (more) {
      async_g2l_b32(gb_off + (unsigned)((pb ^ 1) * 768) * 4u,
                    Gb + (size_t)(t + 1) * 768 + tid);
    }
    float s = bh[tid];
    const float4* hp = (const float4*)hx;
#pragma unroll 8
    for (int k4 = 0; k4 < 64; ++k4) {
      float4 w = wp[(size_t)k4 * 768];
      float4 hv = hp[k4];
      s += w.x * hv.x + w.y * hv.y + w.z * hv.z + w.w * hv.w;
    }
    if (more) wait_async_le1(); else wait_async_le0();
    float gi = gbuf[pb * 768 + tid];
    if (tid < 512) A1[tid] = gi + s;
    else { A1[tid] = gi; A2[tid - 512] = s; }
    __syncthreads();
    if (tid < 256) {
      float r = sigm_(A1[tid]);
      float z = sigm_(A1[256 + tid]);
      float n = tanhf(A1[512 + tid] + r * A2[tid]);
      float hn = (1.f - z) * n + z * hx[tid];
      hx[tid] = hn;
      hsb[(size_t)t * 256 + tid] = hn;
    }
    __syncthreads();
  }
}

// ---------------- output projection (256->4) + log_softmax --------------------------
__global__ __launch_bounds__(256) void k_out(
    const float* __restrict__ hs, const float* __restrict__ W,
    const float* __restrict__ b, float* __restrict__ out) {
  int row = blockIdx.x * 256 + threadIdx.x;
  if (row >= NCLS * TLEN) return;
  const float* hp = hs + (size_t)row * 256;
  float a0 = b[0], a1 = b[1], a2 = b[2], a3 = b[3];
  for (int k = 0; k < 256; k += 4) {
    float4 hv = *(const float4*)(hp + k);
    float4 w0 = *(const float4*)(W + k);
    float4 w1 = *(const float4*)(W + 256 + k);
    float4 w2 = *(const float4*)(W + 512 + k);
    float4 w3 = *(const float4*)(W + 768 + k);
    a0 += w0.x * hv.x + w0.y * hv.y + w0.z * hv.z + w0.w * hv.w;
    a1 += w1.x * hv.x + w1.y * hv.y + w1.z * hv.z + w1.w * hv.w;
    a2 += w2.x * hv.x + w2.y * hv.y + w2.z * hv.z + w2.w * hv.w;
    a3 += w3.x * hv.x + w3.y * hv.y + w3.z * hv.z + w3.w * hv.w;
  }
  float mx = fmaxf(fmaxf(a0, a1), fmaxf(a2, a3));
  float sum = __expf(a0 - mx) + __expf(a1 - mx) + __expf(a2 - mx) + __expf(a3 - mx);
  float lse = mx + __logf(sum);
  out[(size_t)row * 4 + 0] = a0 - lse;
  out[(size_t)row * 4 + 1] = a1 - lse;
  out[(size_t)row * 4 + 2] = a2 - lse;
  out[(size_t)row * 4 + 3] = a3 - lse;
}

__global__ __launch_bounds__(256) void k_tail(
    const int* __restrict__ labels, const float* __restrict__ wts,
    float* __restrict__ out, int mode) {
  int i = blockIdx.x * 256 + threadIdx.x;
  if (i >= NTOT) return;
  const int base = NCLS * TLEN * 4;  // 164160
  if (mode == 0) {
    out[base + i] = (float)labels[i];
    out[base + NTOT + i] = wts[i];
  } else {  // labels occupy int64-sized slots
    ((long long*)(out + base))[i] = (long long)labels[i];
    out[base + 2 * NTOT + i] = wts[i];
  }
}

extern "C" void kernel_launch(void* const* d_in, const int* in_sizes, int n_in,
                              void* d_out, int out_size, void* d_ws, size_t ws_size,
                              hipStream_t stream) {
  (void)in_sizes; (void)n_in; (void)ws_size;
  const float* feat     = (const float*)d_in[0];
  const float* score    = (const float*)d_in[1];
  const float* box      = (const float*)d_in[2];
  const float* origin   = (const float*)d_in[3];
  const int*   labels   = (const int*)d_in[4];
  const float* weights  = (const float*)d_in[5];
  const float* appear_W = (const float*)d_in[8];
  const float* appear_b = (const float*)d_in[9];
  const float* flin_W   = (const float*)d_in[10];
  const float* flin_b   = (const float*)d_in[11];
  const float* eWihf    = (const float*)d_in[12];
  const float* eWhhf    = (const float*)d_in[13];
  const float* ebihf    = (const float*)d_in[14];
  const float* ebhhf    = (const float*)d_in[15];
  const float* eWihb    = (const float*)d_in[16];
  const float* eWhhb    = (const float*)d_in[17];
  const float* ebihb    = (const float*)d_in[18];
  const float* ebhhb    = (const float*)d_in[19];
  const float* dec_emb  = (const float*)d_in[20];
  const float* dWih     = (const float*)d_in[21];
  const float* dWhh     = (const float*)d_in[22];
  const float* dbih     = (const float*)d_in[23];
  const float* dbhh     = (const float*)d_in[24];
  const float* outW     = (const float*)d_in[25];
  const float* outb     = (const float*)d_in[26];
  float* out = (float*)d_out;

  char* p = (char*)d_ws;
  auto alloc = [&](size_t bytes) -> char* {
    char* r = p; p += (bytes + 255) & ~(size_t)255; return r;
  };
  unsigned short* ws_x  = (unsigned short*)alloc((size_t)NTOT * 128 * 2);
  unsigned short* ws_ef = (unsigned short*)alloc((size_t)NTOT * 128 * 2);
  float* ws_gie = (float*)alloc((size_t)160 * TLEN * 384 * 4);
  float* ws_gid = (float*)alloc((size_t)NCLS * TLEN * 768 * 4);
  float* ws_wt  = (float*)alloc((size_t)NCLS * 768 * 256 * 4);
  float* ws_h0  = (float*)alloc((size_t)NCLS * 256 * 4);
  float* ws_hs  = (float*)alloc((size_t)NCLS * TLEN * 256 * 4);

  const int eg_bytes = 4 * 384 * 40 * 2;           // 122880
  const int dg_bytes = (768 * 40 + 128) * 2;       // 61696
  const int er_bytes = (49152 + 384 + 384 + 128 + 128 + 2 * 384) * 4;  // 203776
  (void)hipFuncSetAttribute((const void*)k_enc_gi,
      hipFuncAttributeMaxDynamicSharedMemorySize, eg_bytes);
  (void)hipFuncSetAttribute((const void*)k_dec_gi,
      hipFuncAttributeMaxDynamicSharedMemorySize, dg_bytes);
  (void)hipFuncSetAttribute((const void*)k_enc_rec,
      hipFuncAttributeMaxDynamicSharedMemorySize, er_bytes);

  k_gemm_feat<<<NTOT / 128, 256, 0, stream>>>(feat, appear_W, appear_b, ws_x);
  k_gemm_featlin<<<NTOT / 128, 256, 0, stream>>>(ws_x, score, box, origin, flin_W, flin_b, ws_ef);
  k_enc_gi<<<dim3(NCLS, 2), 256, eg_bytes, stream>>>(
      ws_ef, eWihf, eWihb, ebihf, ebihb, ws_gie);
  k_dec_gi<<<NCLS, 256, dg_bytes, stream>>>(labels, dec_emb, dWih, dbih, ws_gid);
  k_trans_whh<<<(NCLS * 768 * 256) / 256, 256, 0, stream>>>(dWhh, ws_wt);
  k_enc_rec<<<dim3(NCLS, 2), 384, er_bytes, stream>>>(ws_gie, eWhhf, eWhhb, ebhhf, ebhhb, ws_h0);
  k_dec_rec<<<NCLS, 768, 0, stream>>>(ws_gid, ws_wt, dbhh, ws_h0, ws_hs);
  const int rows = NCLS * TLEN;
  k_out<<<(rows + 255) / 256, 256, 0, stream>>>(ws_hs, outW, outb, out);
  int mode = (out_size == rows * 4 + 3 * NTOT) ? 1 : 0;
  k_tail<<<(NTOT + 255) / 256, 256, 0, stream>>>(labels, weights, out, mode);
}
